// ConvRec_12266426597624
// MI455X (gfx1250) — matrix operand order repaired and checked by reading, not compiled, not run on hardware
//
#include <hip/hip_runtime.h>
#include <hip/hip_bf16.h>

typedef __attribute__((ext_vector_type(16))) _Float16 v16h;
typedef __attribute__((ext_vector_type(8)))  float    v8f;
typedef __attribute__((ext_vector_type(4)))  _Float16 h4;
typedef unsigned u32x4 __attribute__((ext_vector_type(4)));
typedef int      i32x8 __attribute__((ext_vector_type(8)));
typedef int      i32x4 __attribute__((ext_vector_type(4)));

// Model constants (from reference)
constexpr int Lc = 2, Hc = 4, Kc = 5, Cc = 128, Fc = 512, Bc = 512, Tc = 200;
constexpr int Mrows = Bc * Tc;   // 102400 (multiple of 128)
constexpr int HK = Hc * Kc;      // 20
#define LN_EPS 1e-5f

#if defined(__AMDGCN__) && __has_builtin(__builtin_amdgcn_tensor_load_to_lds)
#define HAVE_TDM 1
#endif

// ---------------------------------------------------------------------------
// WMMA fragment loaders (CDNA5 16-bit layouts, CDNA5 ISA §7.12.2)
// A (16x32, MxK): lane l<16 -> row M=l, K = {k0..k0+7, k0+16..k0+23}
//                 lane l>=16 -> row M=l-16,  K = {k0+8..k0+15, k0+24..k0+31}
// B (32x16, KxN) from transposed storage BT[n][k]: same K split as A
//                 for column N = lane&15.
// ---------------------------------------------------------------------------
union FragU { v16h v; uint4 q[2]; };

__device__ inline v16h load_a_frag(const _Float16* A, int lda, int row, int khalf, int k0) {
    FragU f;
    const _Float16* p = A + (size_t)row * lda + k0 + khalf * 8;
    f.q[0] = *(const uint4*)p;
    f.q[1] = *(const uint4*)(p + 16);
    return f.v;
}
// B fragment out of the LDS-staged tile (row pitch = KD+8 halves, conflict-free)
__device__ inline v16h load_bt_lds(const _Float16* sm, int pitch, int col, int khalf, int k0) {
    FragU f;
    const _Float16* p = sm + (size_t)col * pitch + k0 + khalf * 8;
    f.q[0] = *(const uint4*)p;
    f.q[1] = *(const uint4*)(p + 16);
    return f.v;
}

// ---------------------------------------------------------------------------
// Stage a (NCOLS x KD) f16 weight tile BT[n0 .. n0+NCOLS)[0..KD) into LDS with
// row pitch KD+8 halves. Preferred path: Tensor Data Mover (2D D#, pad_enable
// inserts 4 dwords per KD*2-byte row => pitch KD+8). Fallback: cooperative copy.
// ---------------------------------------------------------------------------
template<int KD, int NCOLS>
__device__ inline void stage_bt(const _Float16* BT, int n0, int nrows, _Float16* sm) {
    constexpr int PITCH = KD + 8;
#if defined(HAVE_TDM)
    if ((threadIdx.x >> 5) == 0) {            // wave 0 drives the TDM
        unsigned long long ga = (unsigned long long)(const char*)(BT + (size_t)n0 * KD);
        unsigned ldsoff = (unsigned)(size_t)sm;         // low 32 bits = LDS byte addr
        // D# group 0: count=1, lds_addr, global_addr[56:0], type=2 ("image")
        u32x4 g0 = { 1u, ldsoff, (unsigned)ga,
                     (unsigned)((ga >> 32) & 0x01FFFFFFu) | (2u << 30) };
        // D# group 1: data_size=2B, pad_enable, pad_interval = KD*2 bytes,
        // pad_amount = 4 dwords, tensor_dim0=KD, tensor_dim1=nrows,
        // tile_dim0=KD, tile_dim1=NCOLS, dim0_stride=KD
        constexpr unsigned PADI = (KD == 128) ? 5u : 7u;   // (1<<PADI)*8B == KD*2B
        unsigned w0 = (1u << 16) | (1u << 20) | (PADI << 22) | (3u << 25);
        unsigned w1 = ((unsigned)KD & 0xFFFFu) << 16;
        unsigned w2 = ((unsigned)KD >> 16) | (((unsigned)nrows & 0xFFFFu) << 16);
        unsigned w3 = (((unsigned)nrows >> 16) & 0xFFFFu) | ((unsigned)KD << 16);
        unsigned w4 = (unsigned)NCOLS;
        unsigned w5 = (unsigned)KD;
        i32x8 g1 = { (int)w0, (int)w1, (int)w2, (int)w3, (int)w4, (int)w5, 0, 0 };
        i32x4 z4 = { 0, 0, 0, 0 };
#if __clang_major__ >= 23
        i32x8 z8 = { 0, 0, 0, 0, 0, 0, 0, 0 };
        __builtin_amdgcn_tensor_load_to_lds(g0, g1, z4, z4, z8, 0);
#else
        __builtin_amdgcn_tensor_load_to_lds(g0, g1, z4, z4, 0);
#endif
        __builtin_amdgcn_s_wait_tensorcnt(0);
    }
    __syncthreads();
#else
    constexpr int QPR = KD / 8;               // uint4 per row
    for (int i = threadIdx.x; i < NCOLS * QPR; i += 256) {
        int r = i / QPR, cq = i % QPR;
        *(uint4*)(sm + (size_t)r * PITCH + cq * 8) =
            *(const uint4*)(BT + (size_t)(n0 + r) * KD + cq * 8);
    }
    __syncthreads();
#endif
    (void)nrows;
}

__device__ inline float wave_reduce_add(float s) {
#pragma unroll
    for (int m = 16; m >= 1; m >>= 1) s += __shfl_xor(s, m, 32);
    return s;
}

// ---------------------------------------------------------------------------
// Embedding gather: X[t,b,c] = item_emb[seq[b,t], c]   (TBC layout)
// ---------------------------------------------------------------------------
__global__ __launch_bounds__(256) void gather_k(const float* __restrict__ emb,
                                                const int* __restrict__ seq,
                                                float* __restrict__ X) {
    int idx = blockIdx.x * 256 + threadIdx.x;           // over T*B*C
    if (idx >= Tc * Bc * Cc) return;
    int c  = idx & (Cc - 1);
    int tb = idx >> 7;                                   // t*B + b
    int b  = tb & (Bc - 1);
    int t  = tb >> 9;
    int item = seq[b * Tc + t];
    X[idx] = emb[(size_t)item * Cc + c];
}

// ---------------------------------------------------------------------------
// LayerNorm over C=128: one wave per row, float4 per lane, shfl reductions.
// Optionally emits an f16 copy (GEMM A operand).
// ---------------------------------------------------------------------------
__global__ __launch_bounds__(256) void layernorm_k(const float* __restrict__ X,
                                                   float* __restrict__ Out,
                                                   _Float16* __restrict__ OutH,
                                                   const float* __restrict__ g,
                                                   const float* __restrict__ bta,
                                                   int M) {
    int row  = blockIdx.x * 8 + (threadIdx.x >> 5);
    int lane = threadIdx.x & 31;
    if (row >= M) return;
    const float4 v = *(const float4*)(X + (size_t)row * Cc + lane * 4);
    float mean = wave_reduce_add(v.x + v.y + v.z + v.w) * (1.f / Cc);
    float dx = v.x - mean, dy = v.y - mean, dz = v.z - mean, dw = v.w - mean;
    float var = wave_reduce_add(dx * dx + dy * dy + dz * dz + dw * dw) * (1.f / Cc);
    float rinv = rsqrtf(var + LN_EPS);
    float4 gg = *(const float4*)(g + lane * 4);
    float4 bb = *(const float4*)(bta + lane * 4);
    float4 o;
    o.x = dx * rinv * gg.x + bb.x;
    o.y = dy * rinv * gg.y + bb.y;
    o.z = dz * rinv * gg.z + bb.z;
    o.w = dw * rinv * gg.w + bb.w;
    *(float4*)(Out + (size_t)row * Cc + lane * 4) = o;
    if (OutH) {
        h4 hv = { (_Float16)o.x, (_Float16)o.y, (_Float16)o.z, (_Float16)o.w };
        *(h4*)(OutH + (size_t)row * Cc + lane * 4) = hv;
    }
}

// ---------------------------------------------------------------------------
// Conv-weight projection + softmax over K: one thread per (t,b,h).
// ---------------------------------------------------------------------------
__global__ __launch_bounds__(256) void convw_softmax_k(const float* __restrict__ XN,
                                                       const float* __restrict__ cw,
                                                       const float* __restrict__ cb,
                                                       float* __restrict__ Wout) {
    int idx = blockIdx.x * 256 + threadIdx.x;           // over T*B*H
    if (idx >= Tc * Bc * Hc) return;
    int h  = idx & (Hc - 1);
    int tb = idx >> 2;
    const float* x = XN + (size_t)tb * Cc;
    float acc[Kc];
#pragma unroll
    for (int k = 0; k < Kc; ++k) acc[k] = cb[h * Kc + k];
    for (int c = 0; c < Cc; ++c) {
        float xv = x[c];
        const float* wr = cw + (size_t)c * HK + h * Kc;
#pragma unroll
        for (int k = 0; k < Kc; ++k) acc[k] += xv * wr[k];
    }
    float mx = acc[0];
#pragma unroll
    for (int k = 1; k < Kc; ++k) mx = fmaxf(mx, acc[k]);
    float s = 0.f;
#pragma unroll
    for (int k = 0; k < Kc; ++k) { acc[k] = __expf(acc[k] - mx); s += acc[k]; }
    float rs = 1.f / s;
#pragma unroll
    for (int k = 0; k < Kc; ++k) Wout[(size_t)tb * HK + h * Kc + k] = acc[k] * rs;
}

// ---------------------------------------------------------------------------
// Causal dynamic conv: y[t,b,c] = sum_k w[t,b,h(c),k] * xn[t+k-(K-1), b, c]
// ---------------------------------------------------------------------------
__global__ __launch_bounds__(256) void dynconv_k(const float* __restrict__ XN,
                                                 const float* __restrict__ W,
                                                 float* __restrict__ Y) {
    int idx = blockIdx.x * 256 + threadIdx.x;           // over T*B*C
    if (idx >= Tc * Bc * Cc) return;
    int c  = idx & (Cc - 1);
    int tb = idx >> 7;
    int b  = tb & (Bc - 1);
    int t  = tb >> 9;
    int h  = c >> 5;                                     // R = C/H = 32
    const float* wr = W + (size_t)tb * HK + h * Kc;
    float acc = 0.f;
#pragma unroll
    for (int k = 0; k < Kc; ++k) {
        int tt = t + k - (Kc - 1);
        if (tt >= 0) acc += wr[k] * XN[((size_t)tt * Bc + b) * Cc + c];
    }
    Y[idx] = acc;
}

// ---------------------------------------------------------------------------
// Weight pack: f32 -> f16 + transpose so B fragments are K-contiguous.
// W1T[l][n(512)][k(128)] = fc1_w[l][k][n]; W2T[l][n(128)][k(512)] = fc2_w[l][k][n]
// ---------------------------------------------------------------------------
__global__ __launch_bounds__(256) void pack_weights_k(const float* __restrict__ fc1,
                                                      const float* __restrict__ fc2,
                                                      _Float16* __restrict__ W1T,
                                                      _Float16* __restrict__ W2T) {
    int idx = blockIdx.x * 256 + threadIdx.x;           // over L*C*F
    if (idx >= Lc * Cc * Fc) return;
    int l = idx / (Cc * Fc);
    int r = idx % (Cc * Fc);
    int n1 = r / Cc, k1 = r % Cc;
    W1T[idx] = (_Float16)fc1[(size_t)l * Cc * Fc + (size_t)k1 * Fc + n1];
    int n2 = r / Fc, k2 = r % Fc;
    W2T[idx] = (_Float16)fc2[(size_t)l * Fc * Cc + (size_t)k2 * Cc + n2];
}

// ---------------------------------------------------------------------------
// GEMM1: h1 = relu(Y @ fc1_w + b1), M x 512, K=128. B tile staged in LDS (TDM).
// Block: 128 rows x 64 cols; wave: 16x64 (A fragment reused across 4 WMMAs).
// ---------------------------------------------------------------------------
__global__ __launch_bounds__(256) void gemm_ffn1_k(const _Float16* __restrict__ A,
                                                   const _Float16* __restrict__ BT,
                                                   const float* __restrict__ bias,
                                                   _Float16* __restrict__ Out) {
    __shared__ _Float16 sb[64 * (Cc + 8)];               // 17,408 B
    const int n0 = blockIdx.x * 64;
    stage_bt<Cc, 64>(BT, n0, Fc, sb);

    const int lane  = threadIdx.x & 31;
    const int wave  = threadIdx.x >> 5;
    const int m0    = blockIdx.y * 128 + wave * 16;
    const int lr    = lane & 15;
    const int khalf = lane >> 4;
    v8f acc[4];
#pragma unroll
    for (int nt = 0; nt < 4; ++nt) acc[nt] = (v8f){0.f,0.f,0.f,0.f,0.f,0.f,0.f,0.f};
#pragma unroll
    for (int k0 = 0; k0 < Cc; k0 += 32) {
        v16h a = load_a_frag(A, Cc, m0 + lr, khalf, k0);
#pragma unroll
        for (int nt = 0; nt < 4; ++nt) {
            v16h bm = load_bt_lds(sb, Cc + 8, nt * 16 + lr, khalf, k0);
            acc[nt] = __builtin_amdgcn_wmma_f32_16x16x32_f16(
                false, a, false, bm, (short)0, acc[nt], false, false);
        }
    }
#pragma unroll
    for (int nt = 0; nt < 4; ++nt) {
        int col = n0 + nt * 16 + lr;
        float bv = bias[col];
#pragma unroll
        for (int i = 0; i < 8; ++i) {
            int row = m0 + khalf * 8 + i;
            float v = acc[nt][i] + bv;
            Out[(size_t)row * Fc + col] = (_Float16)(v > 0.f ? v : 0.f);
        }
    }
}

// ---------------------------------------------------------------------------
// GEMM2: x_new = h1 @ fc2_w + b2 + y, M x 128, K=512. B tile (32x512) in LDS.
// Block: 128 rows x 32 cols; wave: 16x32 (2 WMMAs per K-step, 16 K-steps).
// ---------------------------------------------------------------------------
__global__ __launch_bounds__(256) void gemm_ffn2_k(const _Float16* __restrict__ A,
                                                   const _Float16* __restrict__ BT,
                                                   const float* __restrict__ bias,
                                                   const float* __restrict__ Yres,
                                                   float* __restrict__ Out) {
    __shared__ _Float16 sb[32 * (Fc + 8)];               // 33,280 B
    const int n0 = blockIdx.x * 32;
    stage_bt<Fc, 32>(BT, n0, Cc, sb);

    const int lane  = threadIdx.x & 31;
    const int wave  = threadIdx.x >> 5;
    const int m0    = blockIdx.y * 128 + wave * 16;
    const int lr    = lane & 15;
    const int khalf = lane >> 4;
    v8f acc[2];
#pragma unroll
    for (int nt = 0; nt < 2; ++nt) acc[nt] = (v8f){0.f,0.f,0.f,0.f,0.f,0.f,0.f,0.f};
    const _Float16* arow = A + (size_t)(m0 + lr) * Fc;
    for (int k0 = 0; k0 < Fc; k0 += 32) {
        if (k0 + 32 < Fc) __builtin_prefetch(arow + k0 + 32, 0, 0);
        v16h a = load_a_frag(A, Fc, m0 + lr, khalf, k0);
#pragma unroll
        for (int nt = 0; nt < 2; ++nt) {
            v16h bm = load_bt_lds(sb, Fc + 8, nt * 16 + lr, khalf, k0);
            acc[nt] = __builtin_amdgcn_wmma_f32_16x16x32_f16(
                false, a, false, bm, (short)0, acc[nt], false, false);
        }
    }
#pragma unroll
    for (int nt = 0; nt < 2; ++nt) {
        int col = n0 + nt * 16 + lr;
        float bv = bias[col];
#pragma unroll
        for (int i = 0; i < 8; ++i) {
            int row = m0 + khalf * 8 + i;
            size_t o = (size_t)row * Cc + col;
            Out[o] = acc[nt][i] + bv + Yres[o];
        }
    }
}

// ---------------------------------------------------------------------------
// Loss: one wave per (t,b). seq_emb row is SE[(t*B+b)*C]; flat index i = b*T+t.
// ---------------------------------------------------------------------------
__global__ __launch_bounds__(256) void loss_k(const float* __restrict__ SE,
                                              const float* __restrict__ emb,
                                              const int* __restrict__ pos,
                                              const int* __restrict__ neg,
                                              float* __restrict__ acc2) {
    int tb   = blockIdx.x * 8 + (threadIdx.x >> 5);
    int lane = threadIdx.x & 31;
    if (tb >= Tc * Bc) return;
    int b = tb & (Bc - 1);
    int t = tb >> 9;
    int i = b * Tc + t;
    const float4 sv = *(const float4*)(SE + (size_t)tb * Cc + lane * 4);
    int p  = pos[i];
    int ng = neg[i];
    const float4 pv = *(const float4*)(emb + (size_t)p  * Cc + lane * 4);
    const float4 nv = *(const float4*)(emb + (size_t)ng * Cc + lane * 4);
    float dp = wave_reduce_add(sv.x * pv.x + sv.y * pv.y + sv.z * pv.z + sv.w * pv.w);
    float dn = wave_reduce_add(sv.x * nv.x + sv.y * nv.y + sv.z * nv.z + sv.w * nv.w);
    if (lane == 0) {
        float sp = 1.f / (1.f + __expf(-dp));
        float sn = 1.f / (1.f + __expf(-dn));
        float per = -__logf(sp + 1e-24f) - __logf(1.f - sn + 1e-24f);
        float mask = (p != 0) ? 1.f : 0.f;
        atomicAdd(&acc2[0], per * mask);
        atomicAdd(&acc2[1], mask);
    }
}

__global__ void init_acc_k(float* acc2) { acc2[0] = 0.f; acc2[1] = 0.f; }
__global__ void finalize_k(const float* acc2, float* out) { out[0] = acc2[0] / acc2[1]; }

// ---------------------------------------------------------------------------
// Host launch
// ---------------------------------------------------------------------------
static inline size_t align_up(size_t x) { return (x + 255) & ~(size_t)255; }

extern "C" void kernel_launch(void* const* d_in, const int* in_sizes, int n_in,
                              void* d_out, int out_size, void* d_ws, size_t ws_size,
                              hipStream_t stream) {
    (void)in_sizes; (void)n_in; (void)out_size; (void)ws_size;
    const float* item_emb = (const float*)d_in[0];
    const float* conv_w   = (const float*)d_in[1];
    const float* conv_b   = (const float*)d_in[2];
    const float* ln_g     = (const float*)d_in[3];
    const float* ln_b     = (const float*)d_in[4];
    const float* fc1_w    = (const float*)d_in[5];
    const float* fc1_b    = (const float*)d_in[6];
    const float* fc2_w    = (const float*)d_in[7];
    const float* fc2_b    = (const float*)d_in[8];
    const float* sln_g    = (const float*)d_in[9];
    const float* sln_b    = (const float*)d_in[10];
    const int*   seq      = (const int*)d_in[11];
    const int*   pos      = (const int*)d_in[12];
    const int*   neg      = (const int*)d_in[13];
    float* out = (float*)d_out;

    // Workspace carve-up
    char* base = (char*)d_ws;
    size_t off = 0;
    float* X  = (float*)(base + off);  off += align_up((size_t)Tc * Bc * Cc * 4);
    float* XN = (float*)(base + off);  off += align_up((size_t)Tc * Bc * Cc * 4);
    float* Y  = (float*)(base + off);  off += align_up((size_t)Tc * Bc * Cc * 4);
    float* W  = (float*)(base + off);  off += align_up((size_t)Tc * Bc * HK * 4);
    _Float16* YH  = (_Float16*)(base + off); off += align_up((size_t)Mrows * Cc * 2);
    _Float16* H1H = (_Float16*)(base + off); off += align_up((size_t)Mrows * Fc * 2);
    _Float16* W1T = (_Float16*)(base + off); off += align_up((size_t)Lc * Fc * Cc * 2);
    _Float16* W2T = (_Float16*)(base + off); off += align_up((size_t)Lc * Cc * Fc * 2);
    float* ACC = (float*)(base + off); off += 256;

    const int nElem  = Tc * Bc * Cc;         // 13,107,200
    const int gElem  = (nElem + 255) / 256;  // 51200
    const int gConv  = (Tc * Bc * Hc + 255) / 256;
    const int gLN    = Mrows / 8;            // 12800
    const int gPack  = (Lc * Cc * Fc + 255) / 256;

    init_acc_k<<<1, 1, 0, stream>>>(ACC);
    pack_weights_k<<<gPack, 256, 0, stream>>>(fc1_w, fc2_w, W1T, W2T);
    gather_k<<<gElem, 256, 0, stream>>>(item_emb, seq, X);

    for (int l = 0; l < Lc; ++l) {
        layernorm_k<<<gLN, 256, 0, stream>>>(X, XN, (_Float16*)nullptr, sln_g, sln_b, Mrows);
        convw_softmax_k<<<gConv, 256, 0, stream>>>(XN, conv_w + (size_t)l * Cc * HK,
                                                   conv_b + (size_t)l * HK, W);
        dynconv_k<<<gElem, 256, 0, stream>>>(XN, W, Y);
        layernorm_k<<<gLN, 256, 0, stream>>>(Y, Y, YH, ln_g + (size_t)l * Cc,
                                             ln_b + (size_t)l * Cc, Mrows);
        gemm_ffn1_k<<<dim3(Fc / 64, Mrows / 128), 256, 0, stream>>>(
            YH, W1T + (size_t)l * Fc * Cc, fc1_b + (size_t)l * Fc, H1H);
        gemm_ffn2_k<<<dim3(Cc / 32, Mrows / 128), 256, 0, stream>>>(
            H1H, W2T + (size_t)l * Cc * Fc, fc2_b + (size_t)l * Cc, Y, X);
    }

    layernorm_k<<<gLN, 256, 0, stream>>>(X, XN, (_Float16*)nullptr, sln_g, sln_b, Mrows);
    loss_k<<<gLN, 256, 0, stream>>>(XN, item_emb, pos, neg, ACC);
    finalize_k<<<1, 1, 0, stream>>>(ACC, out);
}
